// BaseAttention_32744830665455
// MI455X (gfx1250) — compile-verified, matrix operand order checked by reading, not run on hardware
//
#include <hip/hip_runtime.h>
#include <hip/hip_bf16.h>

// ---------------------------------------------------------------------------
// Types for gfx1250 WMMA (wave32): bf16 inputs, f32 accumulate
// ---------------------------------------------------------------------------
typedef __bf16 bf16;
typedef bf16  v8bf  __attribute__((ext_vector_type(8)));
typedef bf16  v16bf __attribute__((ext_vector_type(16)));
typedef float v8f   __attribute__((ext_vector_type(8)));
typedef int   v4i   __attribute__((ext_vector_type(4)));

#define S_LEN 4096
#define D_IN  2048
#define D_OUT 2048
#define N_HEADS 32
#define N_KV    8
#define HD      64
#define D_KV    (N_KV * HD)   // 512

// GEMM tiling
#define BM 128
#define BN 128
#define BK 32
#define LDK 40                // BK + 8 bf16 padding -> 80B row stride, conflict-free

// ---------------------------------------------------------------------------
// gfx1250 async global->LDS copy (ASYNCcnt-tracked), with sync fallback
// Builtin signature (from hipcc diagnostic): 4 args, param0 = v4i* in the
// global (__device__/as1) address space; we cast accordingly.
// ---------------------------------------------------------------------------
#if defined(__has_builtin)
#if __has_builtin(__builtin_amdgcn_global_load_async_to_lds_b128)
#define HAVE_ASYNC_LDS 1
#endif
#if __has_builtin(__builtin_amdgcn_s_wait_asynccnt)
#define WAIT_ASYNC0() __builtin_amdgcn_s_wait_asynccnt(0)
#endif
#endif
#ifndef HAVE_ASYNC_LDS
#define HAVE_ASYNC_LDS 0
#endif
#ifndef WAIT_ASYNC0
#if HAVE_ASYNC_LDS
#define WAIT_ASYNC0() asm volatile("s_wait_asynccnt 0x0" ::: "memory")
#else
#define WAIT_ASYNC0() ((void)0)
#endif
#endif

#if HAVE_ASYNC_LDS
typedef __attribute__((address_space(1))) v4i gl_v4i;   // global (prints as __device__)
typedef __attribute__((address_space(3))) v4i lds_v4i;  // LDS (prints as __shared__)
#endif

// Build a 16-element bf16 A/B fragment from two contiguous 8-element chunks.
// ISA 16-bit A-layout: lane holds K = {kb..kb+7} in elems 0..7 and
// K = {kb+16..kb+23} in elems 8..15 (kb = (lane>>4)*8). B mirrors this when
// fed from the transposed operand.
__device__ __forceinline__ v16bf make_frag(const bf16* lo, const bf16* hi) {
    v8bf a = *(const v8bf*)lo;
    v8bf b = *(const v8bf*)hi;
    v16bf r;
#pragma unroll
    for (int i = 0; i < 8; ++i) { r[i] = a[i]; r[i + 8] = b[i]; }
    return r;
}

__device__ __forceinline__ v8f wmma_bf16(v16bf a, v16bf b, v8f c) {
    return __builtin_amdgcn_wmma_f32_16x16x32_bf16(
        /*neg_a=*/false, a, /*neg_b=*/false, b,
        /*c_mod=*/(short)0, c, /*reuse_a=*/false, /*reuse_b=*/false);
}

// ---------------------------------------------------------------------------
// fp32 -> bf16 elementwise convert (for x)
// ---------------------------------------------------------------------------
__global__ __launch_bounds__(256) void f32_to_bf16_kernel(
    const float* __restrict__ in, bf16* __restrict__ out, int n) {
    int i = (blockIdx.x * 256 + threadIdx.x) * 4;
    if (i + 3 < n) {
        float4 v = *(const float4*)(in + i);
        out[i + 0] = (bf16)v.x;
        out[i + 1] = (bf16)v.y;
        out[i + 2] = (bf16)v.z;
        out[i + 3] = (bf16)v.w;
    }
}

// ---------------------------------------------------------------------------
// Transpose + convert: WT[c][r] = (bf16)W[r][c].  W is [R x C] fp32.
// ---------------------------------------------------------------------------
__global__ __launch_bounds__(256) void transpose_to_bf16_kernel(
    const float* __restrict__ W, bf16* __restrict__ WT, int R, int C) {
    __shared__ float tile[32][33];
    const int c0 = blockIdx.x * 32;
    const int r0 = blockIdx.y * 32;
    const int tx = threadIdx.x & 31;
    const int ty = threadIdx.x >> 5;  // 0..7
#pragma unroll
    for (int i = 0; i < 32; i += 8)
        tile[ty + i][tx] = W[(size_t)(r0 + ty + i) * C + (c0 + tx)];
    __syncthreads();
#pragma unroll
    for (int i = 0; i < 32; i += 8)
        WT[(size_t)(c0 + ty + i) * R + (r0 + tx)] = (bf16)tile[tx][ty + i];
}

// ---------------------------------------------------------------------------
// Generic bf16 GEMM:  C[M,N] = A[M,K] * BT[N,K]^T   (both row-major, bf16)
// 256 threads = 8 waves; wave grid 4(m) x 2(n); each wave owns 32x64 output
// (2x4 grid of 16x16 WMMA tiles). Double-buffered padded LDS, staged with
// GLOBAL_LOAD_ASYNC_TO_LDS_B128 when available (one s_wait_asynccnt +
// barrier per k-iteration, prefetch fully overlapped with WMMA compute).
// Epilogue modes:
//   1: bf16, head-major   out[(n>>6)][m][n&63]      (Q and K layouts)
//   2: bf16, head-transposed out[(n>>6)][n&63][m]   (V^T layout for P*V)
//   3: f32,  out_f[m*N+n] = c + bias[n]             (final projection)
// ---------------------------------------------------------------------------
__global__ __launch_bounds__(256) void gemm_bf16_kernel(
    const bf16* __restrict__ A, const bf16* __restrict__ BT,
    int M, int N, int K, int mode,
    bf16* __restrict__ out_bf, float* __restrict__ out_f,
    const float* __restrict__ bias, int out_seq) {
    __shared__ bf16 As[2][BM * LDK];
    __shared__ bf16 Bs[2][BN * LDK];

    const int tid  = threadIdx.x;
    const int lane = tid & 31;
    const int w    = tid >> 5;    // 0..7
    const int wm   = w >> 1;      // 0..3 -> 32 rows each
    const int wn   = w & 1;       // 0..1 -> 64 cols each
    const int bm   = blockIdx.y * BM;
    const int bn   = blockIdx.x * BN;

    const int l16  = lane & 15;
    const int half = lane >> 4;
    const int kb   = half * 8;

    v8f acc[2][4];
#pragma unroll
    for (int i = 0; i < 2; ++i)
#pragma unroll
        for (int j = 0; j < 4; ++j) acc[i][j] = (v8f)(0.0f);

    // global->LDS staging: each thread copies 2x 8-bf16 chunks per matrix
    const int lr = tid >> 2;          // row 0..63
    const int lc = (tid & 3) * 8;     // col chunk within BK

    const bf16* gA0 = &A [(size_t)(bm + lr)      * K + lc];
    const bf16* gA1 = &A [(size_t)(bm + lr + 64) * K + lc];
    const bf16* gB0 = &BT[(size_t)(bn + lr)      * K + lc];
    const bf16* gB1 = &BT[(size_t)(bn + lr + 64) * K + lc];

#if HAVE_ASYNC_LDS
    auto stage = [&](int buf, int k0) {
        __builtin_amdgcn_global_load_async_to_lds_b128(
            (gl_v4i*)(gA0 + k0), (lds_v4i*)&As[buf][lr * LDK + lc], 0, 0);
        __builtin_amdgcn_global_load_async_to_lds_b128(
            (gl_v4i*)(gA1 + k0), (lds_v4i*)&As[buf][(lr + 64) * LDK + lc], 0, 0);
        __builtin_amdgcn_global_load_async_to_lds_b128(
            (gl_v4i*)(gB0 + k0), (lds_v4i*)&Bs[buf][lr * LDK + lc], 0, 0);
        __builtin_amdgcn_global_load_async_to_lds_b128(
            (gl_v4i*)(gB1 + k0), (lds_v4i*)&Bs[buf][(lr + 64) * LDK + lc], 0, 0);
    };
#else
    auto stage = [&](int buf, int k0) {
        *(v8bf*)&As[buf][lr * LDK + lc]        = *(const v8bf*)(gA0 + k0);
        *(v8bf*)&As[buf][(lr + 64) * LDK + lc] = *(const v8bf*)(gA1 + k0);
        *(v8bf*)&Bs[buf][lr * LDK + lc]        = *(const v8bf*)(gB0 + k0);
        *(v8bf*)&Bs[buf][(lr + 64) * LDK + lc] = *(const v8bf*)(gB1 + k0);
    };
#endif

    stage(0, 0);
    WAIT_ASYNC0();
    __syncthreads();

    const int nk = K / BK;
    for (int kt = 0; kt < nk; ++kt) {
        const int buf = kt & 1;
        // prefetch next tile into the other buffer (overlaps with WMMAs below)
        if (kt + 1 < nk) stage(buf ^ 1, (kt + 1) * BK);

        v16bf af[2], bfrag[4];
#pragma unroll
        for (int i = 0; i < 2; ++i) {
            const bf16* p = &As[buf][(wm * 32 + i * 16 + l16) * LDK + kb];
            af[i] = make_frag(p, p + 16);
        }
#pragma unroll
        for (int j = 0; j < 4; ++j) {
            const bf16* p = &Bs[buf][(wn * 64 + j * 16 + l16) * LDK + kb];
            bfrag[j] = make_frag(p, p + 16);
        }
#pragma unroll
        for (int i = 0; i < 2; ++i)
#pragma unroll
            for (int j = 0; j < 4; ++j)
                acc[i][j] = wmma_bf16(af[i], bfrag[j], acc[i][j]);

        // own async loads complete, then block-wide visibility
        WAIT_ASYNC0();
        __syncthreads();
    }

    // Epilogue. C layout: elem e of v8f, lane -> row = e + 8*half, col = l16.
#pragma unroll
    for (int i = 0; i < 2; ++i) {
#pragma unroll
        for (int j = 0; j < 4; ++j) {
            const int mbase = bm + wm * 32 + i * 16 + half * 8;
            const int ncol  = bn + wn * 64 + j * 16 + l16;
            if (mode == 1) {
                bf16* dst = out_bf + (size_t)(ncol >> 6) * out_seq * HD + (ncol & 63);
#pragma unroll
                for (int e = 0; e < 8; ++e)
                    dst[(size_t)(mbase + e) * HD] = (bf16)acc[i][j][e];
            } else if (mode == 2) {
                bf16* dst = out_bf + ((size_t)(ncol >> 6) * HD + (ncol & 63)) * out_seq + mbase;
#pragma unroll
                for (int e = 0; e < 8; ++e)
                    dst[e] = (bf16)acc[i][j][e];
            } else {
                const float b = bias[ncol];
#pragma unroll
                for (int e = 0; e < 8; ++e)
                    out_f[(size_t)(mbase + e) * N + ncol] = acc[i][j][e] + b;
            }
        }
    }
}

// ---------------------------------------------------------------------------
// Flash attention (causal, GQA). Block = (64-query tile, head); 4 waves,
// each wave owns 16 query rows. Online softmax in exp2 domain.
//   Q  : [N_HEADS][S][64] bf16
//   Kb : [N_KV][S][64]    bf16
//   Vt : [N_KV][64][S]    bf16  (transposed so B-frags are contiguous)
//   ctx: [S][2048]        bf16
// ---------------------------------------------------------------------------
__global__ __launch_bounds__(128) void attn_kernel(
    const bf16* __restrict__ Q, const bf16* __restrict__ Kb,
    const bf16* __restrict__ Vt, bf16* __restrict__ ctx, int S) {
    __shared__ bf16 Pl[4][16 * 40];   // per-wave P staging, padded rows

    const int h    = blockIdx.y;
    const int kvh  = h >> 2;          // GROUP = 4
    const int w    = threadIdx.x >> 5;
    const int lane = threadIdx.x & 31;
    const int l16  = lane & 15;
    const int half = lane >> 4;
    const int kb   = half * 8;
    const int qb   = blockIdx.x * 64 + w * 16;

    const bf16* Qh = Q  + (size_t)h   * S * HD;
    const bf16* Kh = Kb + (size_t)kvh * S * HD;
    const bf16* Vh = Vt + (size_t)kvh * HD * S;

    // Q fragments (head_dim 64 = 2 x K32 fragments), kept in registers
    v16bf aq[2];
    {
        const bf16* qrow = Qh + (size_t)(qb + l16) * HD;
        aq[0] = make_frag(qrow + kb,      qrow + kb + 16);
        aq[1] = make_frag(qrow + 32 + kb, qrow + 32 + kb + 16);
    }

    float mi[8], li[8];
#pragma unroll
    for (int e = 0; e < 8; ++e) { mi[e] = -1e30f; li[e] = 0.0f; }
    v8f accO[4];
#pragma unroll
    for (int j = 0; j < 4; ++j) accO[j] = (v8f)(0.0f);

    const float sc = 0.125f * 1.44269504088896340736f;  // 1/sqrt(64) * log2(e)

    for (int kv0 = 0; kv0 <= qb + 15; kv0 += 32) {
        const bool has1 = (kv0 + 16) <= (qb + 15);

        // prefetch next kv tile (global_prefetch into near caches)
        if (kv0 + 32 <= qb + 15) {
            __builtin_prefetch(Kh + (size_t)(kv0 + 32 + l16) * HD, 0, 3);
            __builtin_prefetch(Vh + (size_t)l16 * S + kv0 + 32, 0, 3);
        }

        // ---- scores: S = Q * K^T (two 16-key tiles, K-dim 64 split in 2)
        v8f s0, s1;
        {
            const bf16* krow = Kh + (size_t)(kv0 + l16) * HD;
            v16bf b0 = make_frag(krow + kb,      krow + kb + 16);
            v16bf b1 = make_frag(krow + 32 + kb, krow + 32 + kb + 16);
            v8f z = (v8f)(0.0f);
            s0 = wmma_bf16(aq[0], b0, z);
            s0 = wmma_bf16(aq[1], b1, s0);
        }
        if (has1) {
            const bf16* krow = Kh + (size_t)(kv0 + 16 + l16) * HD;
            v16bf b0 = make_frag(krow + kb,      krow + kb + 16);
            v16bf b1 = make_frag(krow + 32 + kb, krow + 32 + kb + 16);
            v8f z = (v8f)(0.0f);
            s1 = wmma_bf16(aq[0], b0, z);
            s1 = wmma_bf16(aq[1], b1, s1);
        }

        // ---- scale + causal mask (elementwise cndmask)
#pragma unroll
        for (int e = 0; e < 8; ++e) {
            const int q = qb + e + 8 * half;
            const int k0e = kv0 + l16;
            s0[e] = (k0e > q) ? -1e30f : s0[e] * sc;
            if (has1) s1[e] = (k0e + 16 > q) ? -1e30f : s1[e] * sc;
        }

        // ---- row max across the 16 lanes of each half-wave
        float rt[8];
#pragma unroll
        for (int e = 0; e < 8; ++e) {
            float v = has1 ? fmaxf(s0[e], s1[e]) : s0[e];
            v = fmaxf(v, __shfl_xor(v, 1, 16));
            v = fmaxf(v, __shfl_xor(v, 2, 16));
            v = fmaxf(v, __shfl_xor(v, 4, 16));
            v = fmaxf(v, __shfl_xor(v, 8, 16));
            rt[e] = v;
        }

        float alpha[8], p0[8], p1[8];
#pragma unroll
        for (int e = 0; e < 8; ++e) {
            const float mn = fmaxf(mi[e], rt[e]);
            alpha[e] = __builtin_amdgcn_exp2f(mi[e] - mn);
            mi[e] = mn;
            p0[e] = __builtin_amdgcn_exp2f(s0[e] - mn);
            p1[e] = has1 ? __builtin_amdgcn_exp2f(s1[e] - mn) : 0.0f;
        }

        // ---- row sum
#pragma unroll
        for (int e = 0; e < 8; ++e) {
            float v = p0[e] + p1[e];
            v += __shfl_xor(v, 1, 16);
            v += __shfl_xor(v, 2, 16);
            v += __shfl_xor(v, 4, 16);
            v += __shfl_xor(v, 8, 16);
            li[e] = li[e] * alpha[e] + v;
        }

        // ---- rescale running output
#pragma unroll
        for (int j = 0; j < 4; ++j)
#pragma unroll
            for (int e = 0; e < 8; ++e) accO[j][e] *= alpha[e];

        // ---- C-layout -> A-fragment layout for P via wave-private LDS
        bf16* pw = &Pl[w][0];
#pragma unroll
        for (int e = 0; e < 8; ++e) {
            const int row = e + 8 * half;
            pw[row * 40 + l16]      = (bf16)p0[e];
            pw[row * 40 + 16 + l16] = (bf16)p1[e];
        }
        // same-wave DS ops are in-order: no barrier needed
        const bf16* pr = &Pl[w][l16 * 40 + kb];
        v16bf ap = make_frag(pr, pr + 16);

        // ---- accO += P * V  (V^T rows are contiguous over keys)
#pragma unroll
        for (int j = 0; j < 4; ++j) {
            const bf16* vrow = Vh + (size_t)(j * 16 + l16) * S + kv0 + kb;
            v16bf bv = make_frag(vrow, vrow + 16);
            accO[j] = wmma_bf16(ap, bv, accO[j]);
        }
    }

    // ---- finalize: divide by row sum, store ctx[s][h*64+d]
    float inv[8];
#pragma unroll
    for (int e = 0; e < 8; ++e) inv[e] = 1.0f / li[e];
#pragma unroll
    for (int j = 0; j < 4; ++j) {
#pragma unroll
        for (int e = 0; e < 8; ++e) {
            const int q = qb + e + 8 * half;
            const int d = h * HD + j * 16 + l16;
            ctx[(size_t)q * D_OUT + d] = (bf16)(accO[j][e] * inv[e]);
        }
    }
}

// ---------------------------------------------------------------------------
// Host-side orchestration
// ---------------------------------------------------------------------------
extern "C" void kernel_launch(void* const* d_in, const int* in_sizes, int n_in,
                              void* d_out, int out_size, void* d_ws, size_t ws_size,
                              hipStream_t stream) {
    (void)in_sizes; (void)n_in; (void)out_size; (void)ws_size;
    const float* x  = (const float*)d_in[0];
    const float* Wq = (const float*)d_in[1];
    const float* Wk = (const float*)d_in[2];
    const float* Wv = (const float*)d_in[3];
    const float* Wo = (const float*)d_in[4];
    const float* bo = (const float*)d_in[5];
    float* out = (float*)d_out;

    // workspace carve-up (bytes); total ~80 MB
    char* ws = (char*)d_ws;
    size_t off = 0;
    auto take = [&](size_t bytes) { char* p = ws + off; off += (bytes + 255) & ~(size_t)255; return p; };
    bf16* xb  = (bf16*)take((size_t)S_LEN * D_IN  * 2);
    bf16* WqT = (bf16*)take((size_t)D_OUT * D_IN  * 2);
    bf16* WkT = (bf16*)take((size_t)D_KV  * D_IN  * 2);
    bf16* WvT = (bf16*)take((size_t)D_KV  * D_IN  * 2);
    bf16* WoT = (bf16*)take((size_t)D_OUT * D_OUT * 2);
    bf16* Qb  = (bf16*)take((size_t)N_HEADS * S_LEN * HD * 2);
    bf16* Kb  = (bf16*)take((size_t)N_KV    * S_LEN * HD * 2);
    bf16* Vtb = (bf16*)take((size_t)N_KV    * HD * S_LEN * 2);
    bf16* ctx = (bf16*)take((size_t)S_LEN * D_OUT * 2);

    // 1) converts
    f32_to_bf16_kernel<<<(S_LEN * D_IN) / (256 * 4), 256, 0, stream>>>(x, xb, S_LEN * D_IN);
    transpose_to_bf16_kernel<<<dim3(D_OUT / 32, D_IN / 32), 256, 0, stream>>>(Wq, WqT, D_IN, D_OUT);
    transpose_to_bf16_kernel<<<dim3(D_KV  / 32, D_IN / 32), 256, 0, stream>>>(Wk, WkT, D_IN, D_KV);
    transpose_to_bf16_kernel<<<dim3(D_KV  / 32, D_IN / 32), 256, 0, stream>>>(Wv, WvT, D_IN, D_KV);
    transpose_to_bf16_kernel<<<dim3(D_OUT / 32, D_OUT / 32), 256, 0, stream>>>(Wo, WoT, D_OUT, D_OUT);

    // 2) projections (WMMA GEMMs)
    gemm_bf16_kernel<<<dim3(D_OUT / BN, S_LEN / BM), 256, 0, stream>>>(
        xb, WqT, S_LEN, D_OUT, D_IN, 1, Qb, nullptr, nullptr, S_LEN);
    gemm_bf16_kernel<<<dim3(D_KV / BN, S_LEN / BM), 256, 0, stream>>>(
        xb, WkT, S_LEN, D_KV, D_IN, 1, Kb, nullptr, nullptr, S_LEN);
    gemm_bf16_kernel<<<dim3(D_KV / BN, S_LEN / BM), 256, 0, stream>>>(
        xb, WvT, S_LEN, D_KV, D_IN, 2, Vtb, nullptr, nullptr, S_LEN);

    // 3) causal GQA flash attention
    attn_kernel<<<dim3(S_LEN / 64, N_HEADS), 128, 0, stream>>>(Qb, Kb, Vtb, ctx, S_LEN);

    // 4) output projection + bias -> f32 d_out
    gemm_bf16_kernel<<<dim3(D_OUT / BN, S_LEN / BM), 256, 0, stream>>>(
        ctx, WoT, S_LEN, D_OUT, D_OUT, 3, nullptr, out, bo, 0);
}
